// CausalSelfAttention_66821101191605
// MI455X (gfx1250) — matrix operand order repaired and checked by reading, not compiled, not run on hardware
//
#include <hip/hip_runtime.h>
#include <hip/hip_bf16.h>
#include <cstdint>
#include <cstddef>

typedef __bf16 bf16_t;
typedef __bf16 v16bf __attribute__((ext_vector_type(16)));
typedef float  v8f   __attribute__((ext_vector_type(8)));

#define D_MODEL  1024
#define N_HEADS  16
#define HEAD_DIM 64
#define BATCH    4
#define SEQ      2048
#define MTOT     (BATCH*SEQ)   /* 8192 */

// ---------------------------------------------------------------- fragments
union Frag16 { uint4 q[2]; v16bf v; };

// A-matrix fragment (16-bit, 16x32 MxK):
// lane holds row (row0 + lane%16); K chunks: [kb+(lane/16)*8, +8) and [kb+16+(lane/16)*8, +8)
__device__ inline v16bf load_frag_a(const bf16_t* base, int row0, int kb, int ld, int lane) {
  Frag16 f;
  const int r  = row0 + (lane & 15);
  const int k0 = kb + ((lane >> 4) << 3);
  const char* p = (const char*)(base + r * ld + k0);
  f.q[0] = *(const uint4*)(p);
  f.q[1] = *(const uint4*)(p + 32);   // +16 bf16 elements
  return f.v;
}

// B-matrix fragment (16-bit, 32x16 KxN), B stored row-major as [N][K] (i.e. W rows):
// lane holds column n = n0 + lane%16; K chunks: [kb+(lane/16)*8, +8) and [kb+16+(lane/16)*8, +8)
__device__ inline v16bf load_frag_b(const bf16_t* base, int n0, int kb, int ld, int lane) {
  Frag16 f;
  const int r  = n0 + (lane & 15);
  const int k0 = kb + ((lane >> 4) << 3);
  const char* p = (const char*)(base + r * ld + k0);
  f.q[0] = *(const uint4*)(p);
  f.q[1] = *(const uint4*)(p + 32);
  return f.v;
}

// ---------------------------------------------------------------- async copy (CDNA5)
// GLOBAL_LOAD_ASYNC_TO_LDS_B128, GV mode: per-lane 16B memory->LDS, tracked by ASYNCcnt.
// LDS byte address = low 32 bits of the generic pointer (flat LDS aperture rule).
__device__ inline void async_ldsb128(uint32_t lds_addr, const bf16_t* gptr) {
  asm volatile("global_load_async_to_lds_b128 %0, %1, off"
               :: "v"(lds_addr), "v"(gptr) : "memory");
}
__device__ inline void wait_async0() {
  asm volatile("s_wait_asynccnt 0" ::: "memory");
}

// ---------------------------------------------------------------- converts
__global__ void cvt_f32_bf16(const float* __restrict__ src, bf16_t* __restrict__ dst, int n) {
  int i = blockIdx.x * blockDim.x + threadIdx.x;
  if (i < n) dst[i] = (bf16_t)src[i];
}

// ---------------------------------------------------------------- GEMM
// C[M,N] = A[M,K] * W[N,K]^T, bf16 in, fp32 WMMA accumulate.
// Block tile 128x128, K-step 32, double-buffered LDS fed by async global->LDS copies.
// 256 threads = 8 waves (4 along M x 2 along N), each wave owns 32x64 (2x4 accumulators).
#define BM 128
#define BN 128
#define BK 32
#define LDT 48   // bf16 row pitch (96 B, keeps every fragment chunk 16B-aligned)

template<int MODE>   // 0: plain f32 [M,N] output; 1: scatter qkv -> [B,H,T,hd] bf16
__global__ __launch_bounds__(256) void gemm_wmma_bf16(
    const bf16_t* __restrict__ A, const bf16_t* __restrict__ W,
    float* __restrict__ Cout,
    bf16_t* __restrict__ Qb, bf16_t* __restrict__ Kb, bf16_t* __restrict__ Vb,
    int M, int N, int K)
{
  __shared__ __align__(16) bf16_t As[2][BM * LDT];
  __shared__ __align__(16) bf16_t Bs[2][BN * LDT];

  const int tid  = threadIdx.x;
  const int lane = tid & 31;
  const int wave = tid >> 5;
  const int wm   = wave & 3;   // 0..3 along M (32 rows each)
  const int wn   = wave >> 2;  // 0..1 along N (64 cols each)
  const int m0   = blockIdx.x * BM;
  const int n0   = blockIdx.y * BN;

  // per-thread copy slice: 16 bf16 of one row of each tile (2x b128 per matrix)
  const int crow = tid >> 1;           // 0..127
  const int ccs  = (tid & 1) * 16;     // 0 / 16

  v8f acc[2][4];
#pragma unroll
  for (int i = 0; i < 2; ++i)
#pragma unroll
    for (int j = 0; j < 4; ++j)
      acc[i][j] = (v8f){0.f,0.f,0.f,0.f,0.f,0.f,0.f,0.f};

  const int NKB = K / BK;

  // prologue: stream K-block 0 into buffer 0
  {
    const uint32_t la = (uint32_t)(uintptr_t)(&As[0][crow * LDT + ccs]);
    const uint32_t lb = (uint32_t)(uintptr_t)(&Bs[0][crow * LDT + ccs]);
    const bf16_t* ga = A + (size_t)(m0 + crow) * K + ccs;
    const bf16_t* gb = W + (size_t)(n0 + crow) * K + ccs;
    async_ldsb128(la,      ga);
    async_ldsb128(la + 16, ga + 8);
    async_ldsb128(lb,      gb);
    async_ldsb128(lb + 16, gb + 8);
  }

  for (int kb = 0; kb < NKB; ++kb) {
    wait_async0();        // this wave's pending copies (block kb) have landed
    __syncthreads();      // all waves landed; all waves done reading the other buffer

    if (kb + 1 < NKB) {   // stream block kb+1 into the other buffer, overlapping WMMAs
      const int nb = (kb + 1) & 1;
      const int gk = (kb + 1) * BK;
      const uint32_t la = (uint32_t)(uintptr_t)(&As[nb][crow * LDT + ccs]);
      const uint32_t lb = (uint32_t)(uintptr_t)(&Bs[nb][crow * LDT + ccs]);
      const bf16_t* ga = A + (size_t)(m0 + crow) * K + gk + ccs;
      const bf16_t* gb = W + (size_t)(n0 + crow) * K + gk + ccs;
      async_ldsb128(la,      ga);
      async_ldsb128(la + 16, ga + 8);
      async_ldsb128(lb,      gb);
      async_ldsb128(lb + 16, gb + 8);
    }

    const bf16_t* as = &As[kb & 1][0];
    const bf16_t* bs = &Bs[kb & 1][0];
    v16bf af[2];
#pragma unroll
    for (int i = 0; i < 2; ++i)
      af[i] = load_frag_a(as, wm * 32 + i * 16, 0, LDT, lane);
#pragma unroll
    for (int j = 0; j < 4; ++j) {
      v16bf b = load_frag_b(bs, wn * 64 + j * 16, 0, LDT, lane);
#pragma unroll
      for (int i = 0; i < 2; ++i) {
        acc[i][j] = __builtin_amdgcn_wmma_f32_16x16x32_bf16(
            false, af[i], false, b, (short)0, acc[i][j], false, false);
      }
    }
  }

  // C/D layout: VGPR r -> row r + 8*(lane/16), col lane%16
  const int gsel = lane >> 4;
  const float qscale = 0.125f;   // 1/sqrt(HEAD_DIM)
#pragma unroll
  for (int i = 0; i < 2; ++i) {
#pragma unroll
    for (int j = 0; j < 4; ++j) {
      const int mb = m0 + wm * 32 + i * 16 + gsel * 8;
      const int nc = n0 + wn * 64 + j * 16 + (lane & 15);
#pragma unroll
      for (int r = 0; r < 8; ++r) {
        const int m = mb + r;
        const float val = acc[i][j][r];
        if (MODE == 0) {
          Cout[(size_t)m * N + nc] = val;
        } else {
          const int part = nc >> 10;        // 0:q 1:k 2:v
          const int c = nc & 1023;
          const int h = c >> 6, d = c & 63;
          const int b = m >> 11;            // /SEQ
          const int t = m & 2047;
          const size_t off = (((size_t)(b * N_HEADS + h) * SEQ) + t) * HEAD_DIM + d;
          if      (part == 0) Qb[off] = (bf16_t)(val * qscale);
          else if (part == 1) Kb[off] = (bf16_t)val;
          else                Vb[off] = (bf16_t)val;
        }
      }
    }
  }
}

// ---------------------------------------------------------------- attention
// One workgroup (4 waves, 128 thr) per (b*H+h, 64 q-rows). Flash-style online softmax.
#define QBLK 64
#define KBLK 64
#define LDH  72   // 144-byte row pitch

__global__ __launch_bounds__(128) void attn_wmma(
    const bf16_t* __restrict__ Qg, const bf16_t* __restrict__ Kg,
    const bf16_t* __restrict__ Vg, bf16_t* __restrict__ Og)
{
  __shared__ __align__(16) bf16_t Qs[QBLK * LDH];
  __shared__ __align__(16) bf16_t Ks[KBLK * LDH];
  __shared__ __align__(16) bf16_t Vt[HEAD_DIM * LDH];     // transposed [d][key]
  __shared__ __align__(16) bf16_t Ps[4][16 * LDH];        // per-wave P staging

  const int tid  = threadIdx.x;
  const int lane = tid & 31;
  const int wave = tid >> 5;
  const int qb   = blockIdx.x;
  const int bh   = blockIdx.y;
  const size_t base = (size_t)bh * SEQ * HEAD_DIM;
  const int q0 = qb * QBLK;
  const int gsel = lane >> 4;

  { // load Q block 64x64 (q pre-scaled in projection)
    const int row = tid >> 1;
    const int cs  = (tid & 1) * 32;
    const uint4* g = (const uint4*)(Qg + base + (size_t)(q0 + row) * HEAD_DIM + cs);
    uint4* s = (uint4*)(Qs + row * LDH + cs);
#pragma unroll
    for (int u = 0; u < 4; ++u) s[u] = g[u];
  }

  float mrow[8], lrow[8];
  v8f o[4];
#pragma unroll
  for (int r = 0; r < 8; ++r) { mrow[r] = -3.0e38f; lrow[r] = 0.f; }
#pragma unroll
  for (int j = 0; j < 4; ++j) o[j] = (v8f){0.f,0.f,0.f,0.f,0.f,0.f,0.f,0.f};

  for (int kb = 0; kb <= qb; ++kb) {
    __syncthreads();
    { // K tile straight copy [key][d]
      const int row = tid >> 1;
      const int cs  = (tid & 1) * 32;
      const uint4* g = (const uint4*)(Kg + base + (size_t)(kb * KBLK + row) * HEAD_DIM + cs);
      uint4* s = (uint4*)(Ks + row * LDH + cs);
#pragma unroll
      for (int u = 0; u < 4; ++u) s[u] = g[u];
    }
    // V tile transposed -> Vt[d][key]
    for (int e = tid; e < KBLK * HEAD_DIM; e += 128) {
      const int key = e >> 6, d = e & 63;
      Vt[d * LDH + key] = Vg[base + (size_t)(kb * KBLK + key) * HEAD_DIM + d];
    }
    __syncthreads();

    // S = Q * K^T  (wave's 16 q-rows x 64 keys)
    v8f s[4];
#pragma unroll
    for (int j = 0; j < 4; ++j) s[j] = (v8f){0.f,0.f,0.f,0.f,0.f,0.f,0.f,0.f};
#pragma unroll
    for (int kk = 0; kk < HEAD_DIM; kk += 32) {
      v16bf a = load_frag_a(Qs, wave * 16, kk, LDH, lane);
#pragma unroll
      for (int j = 0; j < 4; ++j) {
        v16bf b = load_frag_b(Ks, j * 16, kk, LDH, lane);
        s[j] = __builtin_amdgcn_wmma_f32_16x16x32_bf16(
            false, a, false, b, (short)0, s[j], false, false);
      }
    }

    // causal mask on the diagonal block
    if (kb == qb) {
      const int qrl = wave * 16 + gsel * 8;       // + r -> local q row
#pragma unroll
      for (int j = 0; j < 4; ++j) {
        const int kl = j * 16 + (lane & 15);
#pragma unroll
        for (int r = 0; r < 8; ++r)
          if (kl > qrl + r) s[j][r] = -3.0e38f;
      }
    }

    // online softmax: row lives in 16 lanes of one VGPR (lanes g*16..g*16+15)
    float rsum[8], alpha[8];
#pragma unroll
    for (int r = 0; r < 8; ++r) {
      float m = s[0][r];
#pragma unroll
      for (int j = 1; j < 4; ++j) m = fmaxf(m, s[j][r]);
#pragma unroll
      for (int off = 8; off >= 1; off >>= 1)
        m = fmaxf(m, __shfl_xor(m, off, 32));
      const float mnew = fmaxf(mrow[r], m);
      alpha[r] = __expf(mrow[r] - mnew);
      mrow[r] = mnew;
    }
#pragma unroll
    for (int r = 0; r < 8; ++r) {
      float acc = 0.f;
#pragma unroll
      for (int j = 0; j < 4; ++j) {
        const float p = __expf(s[j][r] - mrow[r]);
        s[j][r] = p;
        acc += p;
      }
#pragma unroll
      for (int off = 8; off >= 1; off >>= 1)
        acc += __shfl_xor(acc, off, 32);
      rsum[r] = acc;
      lrow[r] = lrow[r] * alpha[r] + rsum[r];
    }
    // rescale O
#pragma unroll
    for (int j = 0; j < 4; ++j)
#pragma unroll
      for (int r = 0; r < 8; ++r) o[j][r] *= alpha[r];

    // stage P (bf16) into per-wave LDS tile as [qrow][key]
    bf16_t* pw = &Ps[wave][0];
#pragma unroll
    for (int j = 0; j < 4; ++j)
#pragma unroll
      for (int r = 0; r < 8; ++r)
        pw[(gsel * 8 + r) * LDH + j * 16 + (lane & 15)] = (bf16_t)s[j][r];
    __syncthreads();

    // O += P * V   (A = P 16x64, B = Vt columns)
#pragma unroll
    for (int kk = 0; kk < KBLK; kk += 32) {
      v16bf a = load_frag_a(pw, 0, kk, LDH, lane);
#pragma unroll
      for (int j = 0; j < 4; ++j) {
        v16bf b = load_frag_b(Vt, j * 16, kk, LDH, lane);
        o[j] = __builtin_amdgcn_wmma_f32_16x16x32_bf16(
            false, a, false, b, (short)0, o[j], false, false);
      }
    }
  }

  // normalize and store O as [B*T, D] bf16 (channel = h*64 + d)
  const int h = bh & (N_HEADS - 1);
  const int b = bh >> 4;
  float inv[8];
#pragma unroll
  for (int r = 0; r < 8; ++r) inv[r] = (lrow[r] > 0.f) ? (1.0f / lrow[r]) : 0.f;
#pragma unroll
  for (int j = 0; j < 4; ++j) {
    const int dcol = j * 16 + (lane & 15);
#pragma unroll
    for (int r = 0; r < 8; ++r) {
      const int t = q0 + wave * 16 + gsel * 8 + r;
      Og[((size_t)(b * SEQ + t)) * D_MODEL + h * HEAD_DIM + dcol] = (bf16_t)(o[j][r] * inv[r]);
    }
  }
}

// ---------------------------------------------------------------- launcher
extern "C" void kernel_launch(void* const* d_in, const int* in_sizes, int n_in,
                              void* d_out, int out_size, void* d_ws, size_t ws_size,
                              hipStream_t stream) {
  (void)in_sizes; (void)n_in; (void)out_size; (void)ws_size;
  const float* x    = (const float*)d_in[0];
  const float* Wqkv = (const float*)d_in[1];
  const float* Wout = (const float*)d_in[2];
  float* out = (float*)d_out;

  char* ws = (char*)d_ws;
  size_t off = 0;
  auto carve = [&](size_t bytes) -> void* {
    void* p = ws + off;
    off += (bytes + 255) & ~(size_t)255;
    return p;
  };
  bf16_t* xb    = (bf16_t*)carve((size_t)MTOT * D_MODEL * 2);
  bf16_t* wqkvb = (bf16_t*)carve((size_t)3 * D_MODEL * D_MODEL * 2);
  bf16_t* woutb = (bf16_t*)carve((size_t)D_MODEL * D_MODEL * 2);
  bf16_t* qbuf  = (bf16_t*)carve((size_t)MTOT * D_MODEL * 2);
  bf16_t* kbuf  = (bf16_t*)carve((size_t)MTOT * D_MODEL * 2);
  bf16_t* vbuf  = (bf16_t*)carve((size_t)MTOT * D_MODEL * 2);
  bf16_t* obuf  = (bf16_t*)carve((size_t)MTOT * D_MODEL * 2);

  const int n1 = MTOT * D_MODEL;
  const int n2 = 3 * D_MODEL * D_MODEL;
  const int n3 = D_MODEL * D_MODEL;
  cvt_f32_bf16<<<(n1 + 255) / 256, 256, 0, stream>>>(x, xb, n1);
  cvt_f32_bf16<<<(n2 + 255) / 256, 256, 0, stream>>>(Wqkv, wqkvb, n2);
  cvt_f32_bf16<<<(n3 + 255) / 256, 256, 0, stream>>>(Wout, woutb, n3);

  // qkv projection: [8192,1024] x [3072,1024]^T -> scatter q/k/v
  gemm_wmma_bf16<1><<<dim3(MTOT / BM, (3 * D_MODEL) / BN), 256, 0, stream>>>(
      xb, wqkvb, nullptr, qbuf, kbuf, vbuf, MTOT, 3 * D_MODEL, D_MODEL);

  // causal attention
  attn_wmma<<<dim3(SEQ / QBLK, BATCH * N_HEADS), 128, 0, stream>>>(qbuf, kbuf, vbuf, obuf);

  // output projection: [8192,1024] x [1024,1024]^T -> fp32 out
  gemm_wmma_bf16<0><<<dim3(MTOT / BM, D_MODEL / BN), 256, 0, stream>>>(
      obuf, woutb, out, nullptr, nullptr, nullptr, MTOT, D_MODEL, D_MODEL);
}
